// TransformerBlock_15710990369322
// MI455X (gfx1250) — matrix operand order repaired and checked by reading, not compiled, not run on hardware
//
#include <hip/hip_runtime.h>
#include <hip/hip_bf16.h>
#include <math.h>

#define B_   4
#define N_   2048
#define DP   128
#define DM   512
#define KNN  17
#define BN   (B_ * N_)          // 8192 total points
#define RPB  (N_ * KNN)         // 34816 neighbor-rows per batch

typedef __bf16 bf16;
typedef bf16  v16bf __attribute__((ext_vector_type(16)));
typedef float v8f   __attribute__((ext_vector_type(8)));
typedef float v4f   __attribute__((ext_vector_type(4)));
typedef unsigned int v4u __attribute__((ext_vector_type(4)));
typedef unsigned int v2u __attribute__((ext_vector_type(2)));

// ============================================================================
// KNN: one block = 256 points of one batch; full xyz of the batch in LDS.
// Strict-compare insertion keeps argsort's stable tie order (lower index first).
// ============================================================================
__global__ __launch_bounds__(256)
void knn_kernel(const float* __restrict__ xyz, int* __restrict__ knn)
{
    __shared__ float sx[N_], sy[N_], sz[N_];
    const int b  = blockIdx.x >> 3;          // 8 blocks per batch
    const int p0 = (blockIdx.x & 7) << 8;
    const float* base = xyz + (size_t)b * N_ * 3;
    for (int i = threadIdx.x; i < N_; i += 256) {
        sx[i] = base[3 * i + 0];
        sy[i] = base[3 * i + 1];
        sz[i] = base[3 * i + 2];
    }
    __syncthreads();

    const int p = p0 + threadIdx.x;
    const float px = sx[p], py = sy[p], pz = sz[p];
    float bd[KNN]; int bi[KNN];
#pragma unroll
    for (int j = 0; j < KNN; ++j) { bd[j] = 3.0e38f; bi[j] = 0; }

    for (int m = 0; m < N_; ++m) {
        const float dx = px - sx[m], dy = py - sy[m], dz = pz - sz[m];
        const float d  = dx * dx + dy * dy + dz * dz;
        if (d < bd[KNN - 1]) {
            int j = KNN - 1;
            while (j > 0 && bd[j - 1] > d) { bd[j] = bd[j - 1]; bi[j] = bi[j - 1]; --j; }
            bd[j] = d; bi[j] = m;
        }
    }
    int* o = knn + (size_t)(b * N_ + p) * KNN;
#pragma unroll
    for (int j = 0; j < KNN; ++j) o[j] = bi[j];
}

// ============================================================================
// Weight cast + transpose: in fp32 [Kc][Nc] row-major -> out bf16 [Nc][Kc].
// ============================================================================
__global__ __launch_bounds__(256)
void cast_tr_kernel(const float* __restrict__ in, bf16* __restrict__ out, int Kc, int Nc)
{
    const int i = blockIdx.x * 256 + threadIdx.x;
    if (i >= Kc * Nc) return;
    const int k = i / Nc, n = i - k * Nc;
    out[(size_t)n * Kc + k] = (bf16)in[i];
}

// ============================================================================
// pos-MLP layer 1 (K=3, VALU): pe1[r][c] = relu((xyz[p]-xyz[nb]) . fcd1[:,c]+b)
// One block per neighbor-row; thread owns 4 consecutive channels (vectorized).
// ============================================================================
__global__ __launch_bounds__(128)
void pe1_kernel(const float* __restrict__ xyz_b, const int* __restrict__ knn_b,
                const float* __restrict__ w, const float* __restrict__ bias,
                bf16* __restrict__ pe1)
{
    const int r  = blockIdx.x;                 // 0..RPB-1
    const int p  = r / KNN;
    const int j  = r - p * KNN;
    const int nb = knn_b[p * KNN + j];
    const float dx = xyz_b[3 * p + 0] - xyz_b[3 * nb + 0];
    const float dy = xyz_b[3 * p + 1] - xyz_b[3 * nb + 1];
    const float dz = xyz_b[3 * p + 2] - xyz_b[3 * nb + 2];
    const int c = threadIdx.x * 4;             // 128 threads * 4 = 512 channels
    const v4f w0 = *(const v4f*)(w + c);
    const v4f w1 = *(const v4f*)(w + DM + c);
    const v4f w2 = *(const v4f*)(w + 2 * DM + c);
    const v4f bb = *(const v4f*)(bias + c);
    union { bf16 h[4]; v2u u; } o;
#pragma unroll
    for (int x = 0; x < 4; ++x)
        o.h[x] = (bf16)fmaxf(dx * w0[x] + dy * w1[x] + dz * w2[x] + bb[x], 0.0f);
    *(v2u*)(pe1 + (size_t)r * DM + c) = o.u;
}

// ============================================================================
// WMMA bf16 GEMM, C[M,Nn] = op(A)[M,Kk] @ WT^T + bias (+relu) (+Cadd)
//   AMODE 0: A = fp32 row-major [M][Kk]
//   AMODE 1: A = bf16 row-major [M][Kk]
//   AMODE 2: fused gather row r -> (p=r/17, j=r%17, nb=gidx):
//            A[r][k] = Qf[p][k] - Kf[nb][k] + Pb[r][k]       (q - k + pos_enc)
// WT is the transposed bf16 weight [Nn][Kk] so B tiles load as contiguous rows.
// Block: 128 threads = 4 waves; tile (RW*64)x64; K-step 32; LDS double-buffered;
// global loads for step t+1 pipelined in registers; L2 prefetch for step t+2.
// RW = rows-of-16 per wave (1 or 2): RW=2 -> 8 back-to-back WMMAs per K-step.
// Fragment layouts follow CDNA5 ISA 7.12.2 (16-bit A 16x32 / B 32x16, f32 C/D).
// ============================================================================
#define GB   128     // block threads
#define TN   64
#define TK   32
#define LSTR 40      // LDS row stride in bf16 (80B: padded, 16B aligned)

template <int AMODE, bool RELU, bool OUTBF, bool CADD, int RW>
__global__ __launch_bounds__(GB)
void gemm_wmma(const float* __restrict__ Af, const bf16* __restrict__ Ab,
               const float* __restrict__ Qf, const float* __restrict__ Kf,
               const bf16* __restrict__ Pb, const int* __restrict__ gidx,
               const bf16* __restrict__ WT, const float* __restrict__ bias,
               float* __restrict__ Cf, bf16* __restrict__ Cb,
               const float* __restrict__ Cadd,
               int M, int Nn, int Kk)
{
    constexpr int TMv = RW * 64;
    __shared__ bf16 As[2][TMv * LSTR];
    __shared__ bf16 Bs[2][TN * LSTR];

    const int row0 = blockIdx.x * TMv;
    const int n0   = blockIdx.y * TN;
    const int tid  = threadIdx.x;
    const int wave = tid >> 5;
    const int lane = tid & 31;

    v8f acc[RW][4];
#pragma unroll
    for (int rw = 0; rw < RW; ++rw)
#pragma unroll
        for (int c = 0; c < 4; ++c)
#pragma unroll
            for (int g = 0; g < 8; ++g) acc[rw][c][g] = 0.0f;

    // -------- staged registers (only the active AMODE's arrays survive DCE)
    v4u a_u[RW * 2];                 // AMODE1: 8 bf16 chunks
    v4f a_f[RW * 4];                 // AMODE0: 4 fp32 chunks
    v4f a_q[RW * 4], a_k[RW * 4];    // AMODE2
    v2u a_p[RW * 4];                 // AMODE2: 4 bf16 (pos_enc)
    v4u b_u[2];                      // B: 8 bf16 chunks

    // global -> registers (16B transactions) + L2 prefetch one K-step further
    auto gload = [&](int k0) {
        if constexpr (AMODE == 1) {
#pragma unroll
            for (int i = 0; i < RW * 2; ++i) {
                const int c  = tid + i * GB;           // chunk of 8 bf16
                const int m  = c >> 2;
                const int kk = (c & 3) * 8;
                a_u[i] = *(const v4u*)(Ab + (size_t)(row0 + m) * Kk + k0 + kk);
            }
            __builtin_prefetch(Ab + (size_t)(row0 + (tid >> 2)) * Kk + k0 + TK + (tid & 3) * 8, 0, 1);
        } else if constexpr (AMODE == 0) {
#pragma unroll
            for (int i = 0; i < RW * 4; ++i) {
                const int c  = tid + i * GB;           // chunk of 4 fp32
                const int m  = c >> 3;
                const int kk = (c & 7) * 4;
                a_f[i] = *(const v4f*)(Af + (size_t)(row0 + m) * Kk + k0 + kk);
            }
            __builtin_prefetch(Af + (size_t)(row0 + (tid >> 3)) * Kk + k0 + TK + (tid & 7) * 4, 0, 1);
        } else {
#pragma unroll
            for (int i = 0; i < RW * 4; ++i) {
                const int c  = tid + i * GB;           // chunk of 4 channels
                const int m  = c >> 3;
                const int kk = (c & 7) * 4;
                const int r  = row0 + m;
                const int p  = r / KNN;
                const int j  = r - p * KNN;
                const int nb = gidx[p * KNN + j];
                a_q[i] = *(const v4f*)(Qf + (size_t)p  * DM + k0 + kk);
                a_k[i] = *(const v4f*)(Kf + (size_t)nb * DM + k0 + kk);
                a_p[i] = *(const v2u*)(Pb + (size_t)r  * DM + k0 + kk);
            }
        }
#pragma unroll
        for (int i = 0; i < 2; ++i) {
            const int c  = tid + i * GB;               // chunk of 8 bf16
            const int n  = c >> 2;
            const int kk = (c & 3) * 8;
            b_u[i] = *(const v4u*)(WT + (size_t)(n0 + n) * Kk + k0 + kk);
        }
        __builtin_prefetch(WT + (size_t)(n0 + (tid >> 2)) * Kk + k0 + TK + (tid & 3) * 8, 0, 1);
    };

    // registers -> LDS (b128 / b64 stores, with bf16 conversion / fusion)
    auto lstore = [&](int buf) {
        if constexpr (AMODE == 1) {
#pragma unroll
            for (int i = 0; i < RW * 2; ++i) {
                const int c  = tid + i * GB;
                const int m  = c >> 2;
                const int kk = (c & 3) * 8;
                *(v4u*)(&As[buf][m * LSTR + kk]) = a_u[i];
            }
        } else if constexpr (AMODE == 0) {
#pragma unroll
            for (int i = 0; i < RW * 4; ++i) {
                const int c  = tid + i * GB;
                const int m  = c >> 3;
                const int kk = (c & 7) * 4;
                union { bf16 h[4]; v2u u; } t;
#pragma unroll
                for (int x = 0; x < 4; ++x) t.h[x] = (bf16)a_f[i][x];
                *(v2u*)(&As[buf][m * LSTR + kk]) = t.u;
            }
        } else {
#pragma unroll
            for (int i = 0; i < RW * 4; ++i) {
                const int c  = tid + i * GB;
                const int m  = c >> 3;
                const int kk = (c & 7) * 4;
                union { bf16 h[4]; v2u u; } pw, t;
                pw.u = a_p[i];
#pragma unroll
                for (int x = 0; x < 4; ++x)
                    t.h[x] = (bf16)(a_q[i][x] - a_k[i][x] + (float)pw.h[x]);
                *(v2u*)(&As[buf][m * LSTR + kk]) = t.u;
            }
        }
#pragma unroll
        for (int i = 0; i < 2; ++i) {
            const int c  = tid + i * GB;
            const int n  = c >> 2;
            const int kk = (c & 3) * 8;
            *(v4u*)(&Bs[buf][n * LSTR + kk]) = b_u[i];
        }
    };

    const int nk = Kk / TK;
    const int ao = (lane < 16) ? 0 : 1;                // A half-select (16B units)
    const int bo = (lane < 16) ? 0 : 1;                // B half-select (16B units)

    gload(0);
    for (int t = 0; t < nk; ++t) {
        const int buf = t & 1;
        lstore(buf);
        __syncthreads();
        if (t + 1 < nk) gload((t + 1) * TK);           // overlaps with WMMA below

        // ---- A fragments: lane<16 holds K{0..7,16..23}, lane>=16 K{8..15,24..31}
        union { v16bf v; v4u q[2]; } afr[RW];
#pragma unroll
        for (int rw = 0; rw < RW; ++rw) {
            const int  mrow  = wave * 16 * RW + rw * 16 + (lane & 15);
            const v4u* abase = (const v4u*)(&As[buf][mrow * LSTR]);
            afr[rw].q[0] = abase[ao];
            afr[rw].q[1] = abase[ao + 2];
        }
        // ---- all B fragments, then RW*4 back-to-back WMMAs
        union { v16bf v; v4u q[2]; } bfr[4];
#pragma unroll
        for (int c = 0; c < 4; ++c) {
            const int  ncol  = c * 16 + (lane & 15);
            const v4u* bbase = (const v4u*)(&Bs[buf][ncol * LSTR]);
            bfr[c].q[0] = bbase[bo];
            bfr[c].q[1] = bbase[bo + 2];
        }
#pragma unroll
        for (int rw = 0; rw < RW; ++rw)
#pragma unroll
            for (int c = 0; c < 4; ++c)
                acc[rw][c] = __builtin_amdgcn_wmma_f32_16x16x32_bf16(
                    false, afr[rw].v, false, bfr[c].v, (short)0, acc[rw][c], false, false);
    }

    // ---- epilogue: VGPR g -> M = g (+8 for upper lane half), N = lane&15
#pragma unroll
    for (int rw = 0; rw < RW; ++rw) {
        const int mbase = row0 + wave * 16 * RW + rw * 16 + ((lane < 16) ? 0 : 8);
#pragma unroll
        for (int c = 0; c < 4; ++c) {
            const int col  = n0 + c * 16 + (lane & 15);
            const float bv = bias ? bias[col] : 0.0f;
#pragma unroll
            for (int g = 0; g < 8; ++g) {
                const int rr = mbase + g;
                float val = acc[rw][c][g] + bv;
                if (RELU) val = fmaxf(val, 0.0f);
                if (CADD) val += Cadd[(size_t)rr * Nn + col];
                if (OUTBF) Cb[(size_t)rr * Nn + col] = (bf16)val;
                else       Cf[(size_t)rr * Nn + col] = val;
            }
        }
    }
}

// ============================================================================
// Softmax over the 17 neighbors (per point, per channel) + weighted sum of
// (v[idx] + pos_enc).  One block per point; thread owns 4 channels (float4).
// ============================================================================
__global__ __launch_bounds__(128)
void attn_out_kernel(const float* __restrict__ logits_b, const bf16* __restrict__ pe_b,
                     const float* __restrict__ v_b, const int* __restrict__ knn_b,
                     bf16* __restrict__ resb)
{
    const int p = blockIdx.x;
    const int c = threadIdx.x * 4;               // 128 threads * 4 = 512 channels
    const float scale = 0.044194173824159216f;   // 1/sqrt(512)

    int nbi[KNN];
#pragma unroll
    for (int j = 0; j < KNN; ++j) nbi[j] = knn_b[p * KNN + j];

    v4f l[KNN];
    v4f mx;
#pragma unroll
    for (int x = 0; x < 4; ++x) mx[x] = -3.0e38f;
#pragma unroll
    for (int j = 0; j < KNN; ++j) {
        v4f t = *(const v4f*)(logits_b + (size_t)(p * KNN + j) * DM + c);
#pragma unroll
        for (int x = 0; x < 4; ++x) { t[x] *= scale; mx[x] = fmaxf(mx[x], t[x]); }
        l[j] = t;
    }
    v4f s;
#pragma unroll
    for (int x = 0; x < 4; ++x) s[x] = 0.0f;
#pragma unroll
    for (int j = 0; j < KNN; ++j)
#pragma unroll
        for (int x = 0; x < 4; ++x) { l[j][x] = __expf(l[j][x] - mx[x]); s[x] += l[j][x]; }

    v4f outv;
#pragma unroll
    for (int x = 0; x < 4; ++x) outv[x] = 0.0f;
#pragma unroll
    for (int j = 0; j < KNN; ++j) {
        const v4f vv = *(const v4f*)(v_b + (size_t)nbi[j] * DM + c);
        union { bf16 h[4]; v2u u; } pe;
        pe.u = *(const v2u*)(pe_b + (size_t)(p * KNN + j) * DM + c);
#pragma unroll
        for (int x = 0; x < 4; ++x)
            outv[x] += l[j][x] * (vv[x] + (float)pe.h[x]);
    }
    union { bf16 h[4]; v2u u; } o;
#pragma unroll
    for (int x = 0; x < 4; ++x) o.h[x] = (bf16)(outv[x] / s[x]);
    *(v2u*)(resb + (size_t)p * DM + c) = o.u;
}

// ============================================================================
// Host-side launch
// ============================================================================
extern "C" void kernel_launch(void* const* d_in, const int* in_sizes, int n_in,
                              void* d_out, int out_size, void* d_ws, size_t ws_size,
                              hipStream_t stream)
{
    const float* xyz      = (const float*)d_in[0];
    const float* features = (const float*)d_in[1];
    const float* fc1_w  = (const float*)d_in[2];
    const float* fc1_b  = (const float*)d_in[3];
    const float* fc2_w  = (const float*)d_in[4];
    const float* fc2_b  = (const float*)d_in[5];
    const float* fcd1_w = (const float*)d_in[6];
    const float* fcd1_b = (const float*)d_in[7];
    const float* fcd2_w = (const float*)d_in[8];
    const float* fcd2_b = (const float*)d_in[9];
    const float* fcg1_w = (const float*)d_in[10];
    const float* fcg1_b = (const float*)d_in[11];
    const float* fcg2_w = (const float*)d_in[12];
    const float* fcg2_b = (const float*)d_in[13];
    const float* wq     = (const float*)d_in[14];
    const float* wk     = (const float*)d_in[15];
    const float* wv     = (const float*)d_in[16];
    float* out = (float*)d_out;

    // ---- workspace carving (all offsets 256B aligned) ----
    char* ws = (char*)d_ws;
    size_t off = 0;
    auto carve = [&](size_t bytes) -> void* {
        void* p = ws + off;
        off += (bytes + 255) & ~(size_t)255;
        return p;
    };
    int*  knn    = (int*) carve((size_t)BN * KNN * 4);
    bf16* xb     = (bf16*)carve((size_t)BN * DM * 2);       // x, bf16
    float* q     = (float*)carve((size_t)BN * DM * 4);
    float* xk    = (float*)carve((size_t)BN * DM * 4);
    float* v     = (float*)carve((size_t)BN * DM * 4);
    bf16* fc1_t  = (bf16*)carve((size_t)DP * DM * 2);       // [512][128]
    bf16* fc2_t  = (bf16*)carve((size_t)DM * DP * 2);       // [128][512]
    bf16* fcd2_t = (bf16*)carve((size_t)DM * DM * 2);
    bf16* fcg1_t = (bf16*)carve((size_t)DM * DM * 2);
    bf16* fcg2_t = (bf16*)carve((size_t)DM * DM * 2);
    bf16* wq_t   = (bf16*)carve((size_t)DM * DM * 2);
    bf16* wk_t   = (bf16*)carve((size_t)DM * DM * 2);
    bf16* wv_t   = (bf16*)carve((size_t)DM * DM * 2);
    bf16* P1     = (bf16*)carve((size_t)RPB * DM * 2);      // pe1 -> a1 (reused)
    bf16* P2     = (bf16*)carve((size_t)RPB * DM * 2);      // pos_enc
    float* P3    = (float*)carve((size_t)RPB * DM * 4);     // attn logits (fp32)
    bf16* resb   = (bf16*)carve((size_t)N_ * DM * 2);       // per-batch attn result
    (void)ws_size; (void)in_sizes; (void)n_in; (void)out_size;

    // ---- weight cast+transpose ----
    auto ctr = [&](const float* in, bf16* o, int Kc, int Nc) {
        cast_tr_kernel<<<(Kc * Nc + 255) / 256, 256, 0, stream>>>(in, o, Kc, Nc);
    };
    ctr(fc1_w,  fc1_t,  DP, DM);
    ctr(fc2_w,  fc2_t,  DM, DP);
    ctr(fcd2_w, fcd2_t, DM, DM);
    ctr(fcg1_w, fcg1_t, DM, DM);
    ctr(fcg2_w, fcg2_t, DM, DM);
    ctr(wq,     wq_t,   DM, DM);
    ctr(wk,     wk_t,   DM, DM);
    ctr(wv,     wv_t,   DM, DM);

    // ---- KNN ----
    knn_kernel<<<B_ * 8, 256, 0, stream>>>(xyz, knn);

    // ---- x = features @ fc1 + b  (fp32 A, bf16 out; 128-row tile) ----
    gemm_wmma<0, false, true, false, 2><<<dim3(BN / 128, DM / TN), GB, 0, stream>>>(
        features, nullptr, nullptr, nullptr, nullptr, nullptr,
        fc1_t, fc1_b, nullptr, xb, nullptr, BN, DM, DP);

    // ---- q / k / v projections (bf16 A, fp32 out, no bias; 128-row tile) ----
    gemm_wmma<1, false, false, false, 2><<<dim3(BN / 128, DM / TN), GB, 0, stream>>>(
        nullptr, xb, nullptr, nullptr, nullptr, nullptr,
        wq_t, nullptr, q, nullptr, nullptr, BN, DM, DM);
    gemm_wmma<1, false, false, false, 2><<<dim3(BN / 128, DM / TN), GB, 0, stream>>>(
        nullptr, xb, nullptr, nullptr, nullptr, nullptr,
        wk_t, nullptr, xk, nullptr, nullptr, BN, DM, DM);
    gemm_wmma<1, false, false, false, 2><<<dim3(BN / 128, DM / TN), GB, 0, stream>>>(
        nullptr, xb, nullptr, nullptr, nullptr, nullptr,
        wv_t, nullptr, v, nullptr, nullptr, BN, DM, DM);

    // ---- per-batch neighbor pipeline (keeps intermediates L2-resident) ----
    for (int b = 0; b < B_; ++b) {
        const float* xyz_b = xyz + (size_t)b * N_ * 3;
        const int*   knn_b = knn + (size_t)b * N_ * KNN;
        const float* q_b   = q   + (size_t)b * N_ * DM;
        const float* k_b   = xk  + (size_t)b * N_ * DM;
        const float* v_b   = v   + (size_t)b * N_ * DM;

        // pe1 = relu(diff @ fcd1 + b)   [RPB x 512]  (K=3, VALU)
        pe1_kernel<<<RPB, 128, 0, stream>>>(xyz_b, knn_b, fcd1_w, fcd1_b, P1);

        // pos_enc = pe1 @ fcd2 + b      (bf16 out; 128-row tile)
        gemm_wmma<1, false, true, false, 2><<<dim3(RPB / 128, DM / TN), GB, 0, stream>>>(
            nullptr, P1, nullptr, nullptr, nullptr, nullptr,
            fcd2_t, fcd2_b, nullptr, P2, nullptr, RPB, DM, DM);

        // a1 = relu((q - k[idx] + pos_enc) @ fcg1 + b)  (fused gather A; 64-row)
        gemm_wmma<2, true, true, false, 1><<<dim3(RPB / 64, DM / TN), GB, 0, stream>>>(
            nullptr, nullptr, q_b, k_b, P2, knn_b,
            fcg1_t, fcg1_b, nullptr, P1, nullptr, RPB, DM, DM);

        // logits = a1 @ fcg2 + b        (fp32 out for stable softmax; 128-row)
        gemm_wmma<1, false, false, false, 2><<<dim3(RPB / 128, DM / TN), GB, 0, stream>>>(
            nullptr, P1, nullptr, nullptr, nullptr, nullptr,
            fcg2_t, fcg2_b, P3, nullptr, nullptr, RPB, DM, DM);

        // softmax over neighbors + weighted sum of (v[idx] + pos_enc)
        attn_out_kernel<<<N_, 128, 0, stream>>>(P3, P2, v_b, knn_b, resb);

        // out = res @ fc2 + b + features  (fused residual; 128-row tile)
        gemm_wmma<1, false, false, true, 2><<<dim3(N_ / 128, DP / TN), GB, 0, stream>>>(
            nullptr, resb, nullptr, nullptr, nullptr, nullptr,
            fc2_t, fc2_b, out + (size_t)b * N_ * DP, nullptr,
            features + (size_t)b * N_ * DP, N_, DP, DM);
    }
}
